// TransformerBlock_59253368815815
// MI455X (gfx1250) — matrix operand order repaired and checked by reading, not compiled, not run on hardware
//
#include <hip/hip_runtime.h>
#include <hip/hip_bf16.h>

#define HID    1024
#define NHEADS 16
#define DKH    64

typedef __bf16 bf16_t;
typedef __attribute__((ext_vector_type(16))) __bf16 v16bf;
typedef __attribute__((ext_vector_type(8)))  __bf16 v8bf;
typedef __attribute__((ext_vector_type(8)))  float  v8f;
typedef __attribute__((ext_vector_type(4)))  float  v4f;

static __device__ __forceinline__ v16bf join8(v8bf lo, v8bf hi) {
    v16bf r;
#pragma unroll
    for (int i = 0; i < 8; ++i) { r[i] = lo[i]; r[i + 8] = hi[i]; }
    return r;
}

static __device__ __forceinline__ unsigned lds_addr32(const void* p) {
    // generic shared pointers: aperture lives in the high 32 bits, low 32 = LDS byte address
    return (unsigned)(unsigned long long)p;
}

// async global -> LDS, 16 bytes; INST_OFFSET applies to both LDS and global address
static __device__ __forceinline__ void async_b128(unsigned lds, const void* gaddr) {
    asm volatile("global_load_async_to_lds_b128 %0, %1, off"
                 :: "v"(lds), "v"((unsigned long long)gaddr) : "memory");
}
static __device__ __forceinline__ void async_b128_o16(unsigned lds, const void* gaddr) {
    asm volatile("global_load_async_to_lds_b128 %0, %1, off offset:16"
                 :: "v"(lds), "v"((unsigned long long)gaddr) : "memory");
}

// ---------------- elementwise helpers ----------------

__global__ void cast_bf16_kernel(const float* __restrict__ in, bf16_t* __restrict__ out, size_t n) {
    size_t i = (size_t)blockIdx.x * blockDim.x + threadIdx.x;
    if (i < n) out[i] = (bf16_t)in[i];
}

// out[c*R + r] = in[r*C + c]   (pre-transpose Wq/Wk/Wv so GEMMs are NT)
__global__ void transpose_cast_kernel(const float* __restrict__ in, bf16_t* __restrict__ out, int R, int C) {
    size_t i = (size_t)blockIdx.x * blockDim.x + threadIdx.x;
    size_t n = (size_t)R * C;
    if (i >= n) return;
    int r = (int)(i % R);
    int c = (int)(i / R);
    out[i] = (bf16_t)in[(size_t)r * C + c];
}

// vb [B,T,H,DK] -> vt [B,H,DK,T]
__global__ void vtrans_kernel(const bf16_t* __restrict__ vb, bf16_t* __restrict__ vt, int B, int T) {
    size_t i = (size_t)blockIdx.x * blockDim.x + threadIdx.x;
    size_t n = (size_t)B * T * HID;
    if (i >= n) return;
    int d  = (int)(i % DKH);   size_t rest = i / DKH;
    int h  = (int)(rest % NHEADS); rest /= NHEADS;
    int t  = (int)(rest % T);
    int b  = (int)(rest / T);
    vt[(((size_t)b * NHEADS + h) * DKH + d) * T + t] = vb[i];
}

// ---------------- async-LDS double-buffered WMMA GEMM ----------------
// C[m,n] = alpha * sum_k A[m,k]*Bt[n,k] (+bias) (+relu)
// Block tile 128x128, 8 waves (4m x 2n), wave tile 32x64, k-step 32.

__global__ __launch_bounds__(256)
void gemm_nt_kernel(const bf16_t* __restrict__ A, const bf16_t* __restrict__ Bt,
                    const float* __restrict__ bias, float alpha, int relu,
                    int M, int N, int K,
                    float* __restrict__ Cf, bf16_t* __restrict__ Cb) {
    __shared__ __align__(16) bf16_t sA[2][128 * 32];
    __shared__ __align__(16) bf16_t sB[2][128 * 32];

    const int tid  = threadIdx.x;
    const int lane = tid & 31;
    const int w    = tid >> 5;
    const int wm   = w >> 1;
    const int wn   = w & 1;
    const int half = lane >> 4;
    const int ln   = lane & 15;
    const int m00  = blockIdx.y * 128;
    const int n00  = blockIdx.x * 128;
    const int mw   = wm * 32;
    const int nw   = wn * 64;

    const int crow = tid >> 1;               // 0..127
    const int coff = (tid & 1) * 16;         // element offset in 32-elem row
    const unsigned ldsA0 = lds_addr32(&sA[0][0]) + (unsigned)(crow * 32 + coff) * 2;
    const unsigned ldsB0 = lds_addr32(&sB[0][0]) + (unsigned)(crow * 32 + coff) * 2;
    const unsigned bufStride = 128u * 32u * 2u;

    const bf16_t* gA = A  + (size_t)(m00 + crow) * K + coff;
    const bf16_t* gB = Bt + (size_t)(n00 + crow) * K + coff;

    v8f c[2][4];
#pragma unroll
    for (int s = 0; s < 2; ++s)
#pragma unroll
        for (int nt = 0; nt < 4; ++nt)
            c[s][nt] = (v8f){0.f,0.f,0.f,0.f,0.f,0.f,0.f,0.f};

    auto issue = [&](int buf, int k0) {
        const unsigned la = ldsA0 + (unsigned)buf * bufStride;
        const unsigned lb = ldsB0 + (unsigned)buf * bufStride;
        async_b128    (la, gA + k0);
        async_b128_o16(la, gA + k0);
        async_b128    (lb, gB + k0);
        async_b128_o16(lb, gB + k0);
    };
    auto compute = [&](int buf) {
        const bf16_t* baseA = &sA[buf][0];
        const bf16_t* baseB = &sB[buf][0];
        v16bf af[2];
#pragma unroll
        for (int s = 0; s < 2; ++s) {
            const bf16_t* arow = baseA + (mw + s * 16 + ln) * 32;
            af[s] = join8(*(const v8bf*)(arow + 16 * half),
                          *(const v8bf*)(arow + 16 * half + 8));
        }
#pragma unroll
        for (int nt = 0; nt < 4; ++nt) {
            const bf16_t* brow = baseB + (nw + nt * 16 + ln) * 32 + 16 * half;
            v16bf bf = join8(*(const v8bf*)(brow), *(const v8bf*)(brow + 8));
#pragma unroll
            for (int s = 0; s < 2; ++s)
                c[s][nt] = __builtin_amdgcn_wmma_f32_16x16x32_bf16(false, af[s], false, bf,
                                                                   (short)0, c[s][nt], false, false);
        }
    };

    issue(0, 0);
    int buf = 0;
    for (int k0 = 0; k0 + 32 < K; k0 += 32) {       // steady state: always issue next
        issue(buf ^ 1, k0 + 32);
        asm volatile("s_wait_asynccnt 0x4" ::: "memory");
        __syncthreads();
        compute(buf);
        __syncthreads();
        buf ^= 1;
    }
    asm volatile("s_wait_asynccnt 0x0" ::: "memory");
    __syncthreads();
    compute(buf);

#pragma unroll
    for (int s = 0; s < 2; ++s)
#pragma unroll
        for (int nt = 0; nt < 4; ++nt) {
            int n = n00 + nw + nt * 16 + ln;
            float bv = bias ? bias[n] : 0.f;
#pragma unroll
            for (int r = 0; r < 8; ++r) {
                int mrow = m00 + mw + s * 16 + r + 8 * half;
                float v = alpha * c[s][nt][r] + bv;
                if (relu) v = fmaxf(v, 0.f);
                if (Cf) Cf[(size_t)mrow * N + n] = v;
                if (Cb) Cb[(size_t)mrow * N + n] = (bf16_t)v;
            }
        }
}

// ---------------- Flash attention with async-LDS K/V staging ----------------
// One block = one (b,h); 8 waves handle 8 consecutive 16-query-row tiles and share
// double-buffered K (32x64) and V^T (64x32) tiles in LDS (async loads, ASYNCcnt).
// q,k: [B,T,H,DK] bf16 (q pre-scaled).  vt: [B,H,DK,T] bf16.  mask: [B,1,T,T] f32.
// o: [B,H,T,DK] bf16 (reference's reshape-without-transpose).
// Computes S' = K_tile * q^T so softmaxed P lands directly in WMMA-A layout.

__global__ __launch_bounds__(256)
void attn_kernel(const bf16_t* __restrict__ q, const bf16_t* __restrict__ k,
                 const bf16_t* __restrict__ vt, const float* __restrict__ mask,
                 bf16_t* __restrict__ o, int B, int T) {
    __shared__ __align__(16) bf16_t sK[2][32 * 64];   // [key row][d]
    __shared__ __align__(16) bf16_t sV[2][64 * 32];   // [d][key row]

    const int tid  = threadIdx.x;
    const int lane = tid & 31;
    const int w    = tid >> 5;
    const int mtiles = T >> 4;
    const int chunks = mtiles >> 3;                   // m-tile groups of 8 per (b,h)
    const int bh  = blockIdx.x / chunks;
    const int mt  = (blockIdx.x % chunks) * 8 + w;
    const int h   = bh % NHEADS;
    const int b   = bh / NHEADS;
    const int half = lane >> 4;
    const int ln   = lane & 15;
    const int m    = mt * 16 + ln;

    const size_t kbase  = (size_t)b * T * NHEADS * DKH + (size_t)h * DKH;
    const size_t vtbase = ((size_t)b * NHEADS + h) * DKH * (size_t)T;
    const float* mrow   = mask + ((size_t)b * T + m) * T;

    // q as WMMA-B operand: lane = m, elem e -> d = 16*half + e (+32*chunk)
    const bf16_t* qrow = q + ((size_t)(b * T + m) * NHEADS + h) * DKH;
    v16bf qf[2];
#pragma unroll
    for (int cc = 0; cc < 2; ++cc) {
        const bf16_t* p = qrow + cc * 32 + 16 * half;
        qf[cc] = join8(*(const v8bf*)(p), *(const v8bf*)(p + 8));
    }

    v8f oacc[4];
#pragma unroll
    for (int i = 0; i < 4; ++i) oacc[i] = (v8f){0.f,0.f,0.f,0.f,0.f,0.f,0.f,0.f};
    float rmax = -3.0e38f, rsum = 0.f;

    // copy indexing: K tile 32 rows x 128B (8 thr/row), V tile 64 rows x 64B (4 thr/row)
    const int kcr = tid >> 3, kco = (tid & 7) * 8;    // K: row, elem offset
    const int vcr = tid >> 2, vco = (tid & 3) * 8;    // V: row(d), elem offset
    const unsigned ldsK0 = lds_addr32(&sK[0][0]) + (unsigned)(kcr * 64 + kco) * 2;
    const unsigned ldsV0 = lds_addr32(&sV[0][0]) + (unsigned)(vcr * 32 + vco) * 2;
    const unsigned tileBytes = 32u * 64u * 2u;

    auto issue = [&](int buf, int j) {
        const bf16_t* gk = k  + kbase + (size_t)(j + kcr) * NHEADS * DKH + kco;
        const bf16_t* gv = vt + vtbase + (size_t)vcr * T + j + vco;
        async_b128(ldsK0 + (unsigned)buf * tileBytes, gk);
        async_b128(ldsV0 + (unsigned)buf * tileBytes, gv);
    };

    auto process = [&](int buf, int j) {
        // S' = K_tile(32x64) * q^T : lane = m, elem r -> key col sub*16 + r + 8*half
        v8f s[2];
#pragma unroll
        for (int sub = 0; sub < 2; ++sub) {
            v8f acc = (v8f){0.f,0.f,0.f,0.f,0.f,0.f,0.f,0.f};
#pragma unroll
            for (int cc = 0; cc < 2; ++cc) {
                const bf16_t* kr = &sK[buf][0] + (sub * 16 + ln) * 64 + cc * 32;
                v16bf af = join8(*(const v8bf*)(kr + 16 * half),
                                 *(const v8bf*)(kr + 16 * half + 8));
                acc = __builtin_amdgcn_wmma_f32_16x16x32_bf16(false, af, false, qf[cc],
                                                              (short)0, acc, false, false);
            }
            s[sub] = acc;
        }
#pragma unroll
        for (int sub = 0; sub < 2; ++sub) {
            const float* mp = mrow + j + sub * 16 + 8 * half;
            v4f m0 = *(const v4f*)(mp);
            v4f m1 = *(const v4f*)(mp + 4);
#pragma unroll
            for (int r = 0; r < 4; ++r) { s[sub][r] += m0[r]; s[sub][r + 4] += m1[r]; }
        }
        float tmax = -3.0e38f;
#pragma unroll
        for (int r = 0; r < 8; ++r) tmax = fmaxf(tmax, fmaxf(s[0][r], s[1][r]));
        tmax = fmaxf(tmax, __shfl_xor(tmax, 16, 32));
        float nmax = fmaxf(rmax, tmax);
        float corr = __expf(rmax - nmax);
        rmax = nmax;
        float p0[8], p1[8], tsum = 0.f;
#pragma unroll
        for (int r = 0; r < 8; ++r) {
            p0[r] = __expf(s[0][r] - nmax);
            p1[r] = __expf(s[1][r] - nmax);
            tsum += p0[r] + p1[r];
        }
        tsum += __shfl_xor(tsum, 16, 32);
        rsum = rsum * corr + tsum;
        float cm[8];
#pragma unroll
        for (int r = 0; r < 8; ++r) cm[r] = __shfl(corr, r + 8 * half, 32);
#pragma unroll
        for (int dt = 0; dt < 4; ++dt)
#pragma unroll
            for (int r = 0; r < 8; ++r) oacc[dt][r] *= cm[r];
        // P (16m x 32key) already in WMMA-A layout
        v16bf pa;
#pragma unroll
        for (int e = 0; e < 8; ++e) { pa[e] = (bf16_t)p0[e]; pa[e + 8] = (bf16_t)p1[e]; }
#pragma unroll
        for (int dt = 0; dt < 4; ++dt) {
            const bf16_t* vr = &sV[buf][0] + (dt * 16 + ln) * 32;
            v16bf vf = join8(*(const v8bf*)(vr + 8 * half), *(const v8bf*)(vr + 16 + 8 * half));
            oacc[dt] = __builtin_amdgcn_wmma_f32_16x16x32_bf16(false, pa, false, vf,
                                                               (short)0, oacc[dt], false, false);
        }
    };

    issue(0, 0);
    int buf = 0;
    for (int j = 0; j + 32 < T; j += 32) {
        __builtin_prefetch(mrow + j + 32, 0, 1);      // global_prefetch_b8
        issue(buf ^ 1, j + 32);
        asm volatile("s_wait_asynccnt 0x2" ::: "memory");
        __syncthreads();
        process(buf, j);
        __syncthreads();
        buf ^= 1;
    }
    asm volatile("s_wait_asynccnt 0x0" ::: "memory");
    __syncthreads();
    process(buf, T - 32);

    float inv = 1.f / rsum;
    float im[8];
#pragma unroll
    for (int r = 0; r < 8; ++r) im[r] = __shfl(inv, r + 8 * half, 32);
    const size_t obase = ((size_t)b * NHEADS + h) * T;
#pragma unroll
    for (int dt = 0; dt < 4; ++dt)
#pragma unroll
        for (int r = 0; r < 8; ++r) {
            int trow = mt * 16 + r + 8 * half;
            o[(obase + trow) * DKH + dt * 16 + ln] = (bf16_t)(oacc[dt][r] * im[r]);
        }
}

// ---------------- fused residual + LayerNorm (one block per row, D=1024) ----------------

__global__ __launch_bounds__(256)
void ln_kernel(const float* __restrict__ a, const float* __restrict__ b,
               const float* __restrict__ g, const float* __restrict__ be,
               float* __restrict__ outf, bf16_t* __restrict__ outb, int D) {
    const size_t base = (size_t)blockIdx.x * D;
    float vals[4];
    float s = 0.f, ss = 0.f;
#pragma unroll
    for (int i = 0; i < 4; ++i) {
        int idx = threadIdx.x + i * 256;
        float x = a[base + idx] + (b ? b[base + idx] : 0.f);
        vals[i] = x; s += x; ss += x * x;
    }
#pragma unroll
    for (int off = 16; off >= 1; off >>= 1) {
        s  += __shfl_xor(s,  off, 32);
        ss += __shfl_xor(ss, off, 32);
    }
    __shared__ float sh[16];
    int wid = threadIdx.x >> 5;
    if ((threadIdx.x & 31) == 0) { sh[wid] = s; sh[8 + wid] = ss; }
    __syncthreads();
    s = 0.f; ss = 0.f;
#pragma unroll
    for (int i = 0; i < 8; ++i) { s += sh[i]; ss += sh[8 + i]; }
    float mu  = s / D;
    float var = ss / D - mu * mu;
    float rs  = rsqrtf(var + 1e-5f);
#pragma unroll
    for (int i = 0; i < 4; ++i) {
        int idx = threadIdx.x + i * 256;
        float y = (vals[i] - mu) * rs * g[idx] + be[idx];
        if (outf) outf[base + idx] = y;
        if (outb) outb[base + idx] = (bf16_t)y;
    }
}

// ---------------- host orchestration ----------------

extern "C" void kernel_launch(void* const* d_in, const int* in_sizes, int n_in,
                              void* d_out, int out_size, void* d_ws, size_t ws_size,
                              hipStream_t stream) {
    const float* x    = (const float*)d_in[0];
    const float* mask = (const float*)d_in[1];
    const float* Wq   = (const float*)d_in[2];
    const float* Wk   = (const float*)d_in[3];
    const float* Wv   = (const float*)d_in[4];
    const float* Wo   = (const float*)d_in[5];
    const float* W1   = (const float*)d_in[6];
    const float* b1   = (const float*)d_in[7];
    const float* W2   = (const float*)d_in[8];
    const float* b2   = (const float*)d_in[9];
    const float* g1   = (const float*)d_in[10];
    const float* be1  = (const float*)d_in[11];
    const float* g2   = (const float*)d_in[12];
    const float* be2  = (const float*)d_in[13];

    const int D  = HID;
    const int M  = in_sizes[0] / D;            // B*T
    const int T  = in_sizes[1] / M;
    const int B  = M / T;
    const int D2 = 2 * D;

    char* p = (char*)d_ws;
    auto alloc = [&](size_t bytes) -> void* {
        void* r = (void*)p;
        p += (bytes + 255) & ~(size_t)255;
        return r;
    };
    bf16_t* xb   = (bf16_t*)alloc((size_t)M * D * 2);
    bf16_t* WqT  = (bf16_t*)alloc((size_t)D * D * 2);
    bf16_t* WkT  = (bf16_t*)alloc((size_t)D * D * 2);
    bf16_t* WvT  = (bf16_t*)alloc((size_t)D * D * 2);
    bf16_t* Wob  = (bf16_t*)alloc((size_t)D * D * 2);
    bf16_t* W1b  = (bf16_t*)alloc((size_t)D2 * D * 2);
    bf16_t* W2b  = (bf16_t*)alloc((size_t)D * D2 * 2);
    bf16_t* qb   = (bf16_t*)alloc((size_t)M * D * 2);
    bf16_t* kb   = (bf16_t*)alloc((size_t)M * D * 2);   // contiguous after qb (hb alias)
    bf16_t* vb   = (bf16_t*)alloc((size_t)M * D * 2);
    bf16_t* vt   = (bf16_t*)alloc((size_t)M * D * 2);
    bf16_t* ob   = (bf16_t*)alloc((size_t)M * D * 2);
    float*  attnf= (float*)alloc((size_t)M * D * 4);
    float*  x1f  = (float*)alloc((size_t)M * D * 4);
    bf16_t* x1b  = (bf16_t*)alloc((size_t)M * D * 2);
    bf16_t* hb   = qb;                                  // alias: q/k dead after attention
    float*  fbuf = attnf;                               // alias: attnf dead after LN1

    auto cdiv = [](size_t a, size_t b) { return (a + b - 1) / b; };

    cast_bf16_kernel<<<cdiv((size_t)M * D, 256), 256, 0, stream>>>(x, xb, (size_t)M * D);
    transpose_cast_kernel<<<cdiv((size_t)D * D, 256), 256, 0, stream>>>(Wq, WqT, D, D);
    transpose_cast_kernel<<<cdiv((size_t)D * D, 256), 256, 0, stream>>>(Wk, WkT, D, D);
    transpose_cast_kernel<<<cdiv((size_t)D * D, 256), 256, 0, stream>>>(Wv, WvT, D, D);
    cast_bf16_kernel<<<cdiv((size_t)D * D, 256), 256, 0, stream>>>(Wo, Wob, (size_t)D * D);
    cast_bf16_kernel<<<cdiv((size_t)D2 * D, 256), 256, 0, stream>>>(W1, W1b, (size_t)D2 * D);
    cast_bf16_kernel<<<cdiv((size_t)D * D2, 256), 256, 0, stream>>>(W2, W2b, (size_t)D * D2);

    // QKV projections (q pre-scaled by 1/sqrt(DK))
    dim3 gD(D / 128, M / 128);
    gemm_nt_kernel<<<gD, 256, 0, stream>>>(xb, WqT, nullptr, 0.125f, 0, M, D, D, nullptr, qb);
    gemm_nt_kernel<<<gD, 256, 0, stream>>>(xb, WkT, nullptr, 1.0f,   0, M, D, D, nullptr, kb);
    gemm_nt_kernel<<<gD, 256, 0, stream>>>(xb, WvT, nullptr, 1.0f,   0, M, D, D, nullptr, vb);
    vtrans_kernel<<<cdiv((size_t)M * D, 256), 256, 0, stream>>>(vb, vt, B, T);

    // flash attention: one block per (b, h, 8 m-tiles)
    int blocks = B * NHEADS * ((T / 16) / 8);
    attn_kernel<<<blocks, 256, 0, stream>>>(qb, kb, vt, mask, ob, B, T);

    // output projection: attn = o_flat @ Wo^T
    gemm_nt_kernel<<<gD, 256, 0, stream>>>(ob, Wob, nullptr, 1.0f, 0, M, D, D, attnf, nullptr);

    // x1 = LN(x + attn)
    ln_kernel<<<M, 256, 0, stream>>>(x, attnf, g1, be1, x1f, x1b, D);

    // FFN
    dim3 gH(D2 / 128, M / 128);
    gemm_nt_kernel<<<gH, 256, 0, stream>>>(x1b, W1b, b1, 1.0f, 1, M, D2, D, nullptr, hb);
    gemm_nt_kernel<<<gD, 256, 0, stream>>>(hb, W2b, b2, 1.0f, 0, M, D, D2, fbuf, nullptr);

    // out = LN(x1 + f)
    ln_kernel<<<M, 256, 0, stream>>>(x1f, fbuf, g2, be2, (float*)d_out, nullptr, D);
}
